// MSIW_14113262534678
// MI455X (gfx1250) — compile-verified, matrix operand order checked by reading, not run on hardware
//
#include <hip/hip_runtime.h>
#include <hip/hip_bf16.h>

typedef __attribute__((ext_vector_type(16))) _Float16 v16h;
typedef __attribute__((ext_vector_type(8)))  float    v8f;

#define N_  4
#define C_  19
#define H_  512
#define W_  1024
#define HW_ (H_ * W_)            // 524288 = 2^19
#define NPIX (N_ * HW_)          // 2097152
#define IW_ 0.2f

// ---------------------------------------------------------------------------
// Init: zero the 2*C_ workspace accumulators (hist[0..18], acc[19..37]).
// Runs every launch -> deterministic across graph replays.
// ---------------------------------------------------------------------------
__global__ void msiw_init(float* __restrict__ ws) {
    int t = threadIdx.x;
    if (t < 2 * C_) ws[t] = 0.0f;
}

// ---------------------------------------------------------------------------
// Pass 1 (bandwidth-bound): one full read of the tensor via float4 loads.
// Each thread handles 4 consecutive pixels; per pixel: argmax over C=19,
// s = (sum exp(2(x-m))) / (sum exp(x-m))^2, accumulated into per-class bins.
// LDS atomics per block, then two global f32 atomicAdds (count, sum) per class per block.
// ---------------------------------------------------------------------------
__global__ __launch_bounds__(256) void msiw_pass1(const float* __restrict__ x,
                                                  float* __restrict__ ws) {
    __shared__ unsigned int s_hist[C_];
    __shared__ float        s_acc[C_];
    const int t = threadIdx.x;
    if (t < C_) { s_hist[t] = 0u; s_acc[t] = 0.0f; }
    __syncthreads();

    const int p4 = blockIdx.x * blockDim.x + t;      // quad-pixel index
    const int p  = p4 << 2;                          // first pixel of the quad
    if (p < NPIX) {
        const int n  = p >> 19;                      // HW_ = 2^19
        const int hw = p & (HW_ - 1);                // quad never crosses a channel
        const float4* base =
            (const float4*)(x + (size_t)n * C_ * HW_ + (size_t)hw);

        float4 v[C_];
        #pragma unroll
        for (int c = 0; c < C_; ++c)
            v[c] = base[(size_t)c * (HW_ / 4)];      // global_load_b128, coalesced

        const float* vf = (const float*)v;           // vf[c*4 + j]
        #pragma unroll
        for (int j = 0; j < 4; ++j) {
            float m = vf[j];
            int pred = 0;
            #pragma unroll
            for (int c = 1; c < C_; ++c) {
                float xv = vf[c * 4 + j];
                if (xv > m) { m = xv; pred = c; }
            }
            float S = 0.0f, S2 = 0.0f;
            #pragma unroll
            for (int c = 0; c < C_; ++c) {
                float e = __expf(vf[c * 4 + j] - m); // v_exp_f32
                S  += e;
                S2 += e * e;                         // exp(2(x-m))
            }
            float s = S2 / (S * S);                  // sum_c prob_c^2
            atomicAdd(&s_hist[pred], 1u);
            atomicAdd(&s_acc[pred], s);
        }
    }
    __syncthreads();
    if (t < C_) {
        atomicAdd(&ws[t],       (float)s_hist[t]);   // exact: integer counts < 2^24
        atomicAdd(&ws[C_ + t],  s_acc[t]);
    }
}

// ---------------------------------------------------------------------------
// Pass 2 (single wave32): den_c = max(hist^0.2 * Np^0.8, 1), val_c = acc_c/den_c,
// then reduce the 19 vals with a WMMA ones-matrix row-sum:
//   D = A x Ones, A[m][0] = val_m, A[m][1] = val_{m+16}  (f16 A, f32 accum)
// D rows are the per-class values; fold 8 D VGPRs + shfl_xor(16) -> scalar.
// EXEC is all-1s here (exactly 32 threads, no divergence at the WMMA).
// ---------------------------------------------------------------------------
__global__ __launch_bounds__(32) void msiw_pass2(const float* __restrict__ ws,
                                                 float* __restrict__ out) {
    const int lane = threadIdx.x;
    const float scale = powf((float)NPIX, 1.0f - IW_); // Np^0.8

    float vA = 0.0f, vB = 0.0f;
    if (lane < 16) {
        {   // class c = lane  (always < 19)
            float h   = ws[lane];
            float den = fmaxf(powf(h, IW_) * scale, 1.0f);
            vA = ws[C_ + lane] / den;
        }
        int c2 = lane + 16;
        if (c2 < C_) {                                  // classes 16..18
            float h   = ws[c2];
            float den = fmaxf(powf(h, IW_) * scale, 1.0f);
            vB = ws[C_ + c2] / den;
        }
    }

    // A: 16x32 f16. Lanes 0-15 hold row M=lane; A[0]=K0, A[1]=K1. Rest zero.
    v16h A = {};
    A[0] = (_Float16)vA;
    A[1] = (_Float16)vB;

    // B: 32x16 all-ones -> D[m][n] = sum_k A[m][k] (row sums in every column).
    v16h B;
    #pragma unroll
    for (int i = 0; i < 16; ++i) B[i] = (_Float16)1.0f;

    v8f Cacc = {};
    // emits v_wmma_f32_16x16x32_f16
    v8f D = __builtin_amdgcn_wmma_f32_16x16x32_f16(
        /*neg_a=*/false, A, /*neg_b=*/false, B,
        /*c_mod=*/(short)0, Cacc, /*reuse_a=*/false, /*reuse_b=*/false);

    // Lanes 0-15 hold rows 0-7 in D[0..7]; lanes 16-31 hold rows 8-15.
    float partial = 0.0f;
    #pragma unroll
    for (int i = 0; i < 8; ++i) partial += D[i];
    float total = partial + __shfl_xor(partial, 16, 32);

    if (lane == 0) out[0] = -total / (float)(N_ * C_);
}

// ---------------------------------------------------------------------------
extern "C" void kernel_launch(void* const* d_in, const int* in_sizes, int n_in,
                              void* d_out, int out_size, void* d_ws, size_t ws_size,
                              hipStream_t stream) {
    const float* x   = (const float*)d_in[0];
    float*       out = (float*)d_out;
    float*       ws  = (float*)d_ws;

    msiw_init<<<1, 64, 0, stream>>>(ws);

    const int quads  = NPIX / 4;                 // 524288
    const int block  = 256;
    const int grid   = (quads + block - 1) / block; // 2048
    msiw_pass1<<<grid, block, 0, stream>>>(x, ws);

    msiw_pass2<<<1, 32, 0, stream>>>(ws, out);
}
